// VQEmbed_42631845380237
// MI455X (gfx1250) — matrix operand order repaired and checked by reading, not compiled, not run on hardware
//
#include <hip/hip_runtime.h>
#include <hip/hip_bf16.h>

// ---------------------------------------------------------------------------
// VQ-VAE forward for MI455X (gfx1250, wave32, WMMA + async/TDM staging).
// B=8 T=2048 F=1024 D=256 K=8192.  Dominant cost: 16384x8192x256 dist GEMM
// (68.7 GFLOP) -> bf16 V_WMMA_F32_16X16X32_BF16 with f32 accumulation.
// Tile copies go through GLOBAL_LOAD_ASYNC_TO_LDS / TENSOR_LOAD_TO_LDS when
// the toolchain exposes them (guarded; sync fallback keeps compile safe).
// ---------------------------------------------------------------------------

typedef __bf16 bf16_t;
typedef __attribute__((ext_vector_type(8)))  __bf16 v8bf;
typedef __attribute__((ext_vector_type(16))) __bf16 v16bf;
typedef __attribute__((ext_vector_type(8)))  float  v8f;

#define WMMA_BF16(A, B, C)                                                    \
  __builtin_amdgcn_wmma_f32_16x16x32_bf16(false, (A), false, (B), (short)0,   \
                                          (C), false, false)

static constexpr int BT   = 8 * 2048;   // M = 16384 tokens
static constexpr int FDIM = 1024;
static constexpr int DDIM = 256;
static constexpr int KCB  = 8192;

// ---- CDNA5 async / TDM capability probes -----------------------------------
#if defined(__has_builtin)
#if __has_builtin(__builtin_amdgcn_global_load_async_to_lds_b128) && \
    __has_builtin(__builtin_amdgcn_s_wait_asynccnt)
#define HAVE_ASYNC 1
#else
#define HAVE_ASYNC 0
#endif
#if __has_builtin(__builtin_amdgcn_tensor_load_to_lds) && \
    __has_builtin(__builtin_amdgcn_s_wait_tensorcnt)
#define HAVE_TDM 1
#else
#define HAVE_TDM 0
#endif
#else
#define HAVE_ASYNC 0
#define HAVE_TDM 0
#endif

// 16B global->LDS copy: async (ASYNCcnt) when available, else VGPR roundtrip.
// Builtin prototype (from hipcc diagnostic): params are pointers to
// int __attribute__((vector_size(16))) -- generic, typed pointers.
__device__ __forceinline__ void cp16_lds(bf16_t* dst, const bf16_t* src) {
#if HAVE_ASYNC
  typedef int v4i __attribute__((vector_size(16)));
  __builtin_amdgcn_global_load_async_to_lds_b128((v4i*)src, (v4i*)dst, 0, 0);
#else
  *(v8bf*)dst = *(const v8bf*)src;
#endif
}

__device__ __forceinline__ void cp_fence() {
#if HAVE_ASYNC
  __builtin_amdgcn_s_wait_asynccnt(0);
#endif
}

#if HAVE_TDM
typedef unsigned int u32x4 __attribute__((ext_vector_type(4)));
typedef int i32x4 __attribute__((ext_vector_type(4)));
typedef int i32x8 __attribute__((ext_vector_type(8)));

// 2D TDM load: tile 32 x 256 bf16 elems, row stride 256 elems, writing LDS
// rows of 64B followed by 16B padding (pad_interval=16 DW, pad_amount=4 DW)
// -> exactly our 80B (LDT=40) padded LDS layout.
__device__ __forceinline__ void tdm_load_btile(unsigned lds_off,
                                               const void* gaddr) {
  unsigned long long ga = (unsigned long long)gaddr;
  u32x4 g0 = {0u, 0u, 0u, 0u};
  g0[0] = 1u;                                    // count=1 (valid descriptor)
  g0[1] = lds_off;                               // lds_addr (bytes)
  g0[2] = (unsigned)ga;                          // global_addr[31:0]
  g0[3] = (unsigned)((ga >> 32) & 0x01FFFFFFu)   // global_addr[56:32]
          | (2u << 30);                          // type=2 ("image")
  i32x8 g1 = {0, 0, 0, 0, 0, 0, 0, 0};
  g1[0] = (1 << 16)      // data_size = 2B
        | (1 << 20)      // pad_enable
        | (3 << 22)      // pad_interval: 16 DWORDs (64B)
        | (3 << 25);     // pad_amount: 4 DWORDs (16B)
  g1[1] = 32 << 16;      // tensor_dim0 = 32 (tile == tensor: no OOB)
  g1[2] = 256 << 16;     // tensor_dim1 = 256
  g1[3] = 32 << 16;      // tile_dim0 = 32 elems (64B rows)
  g1[4] = 256;           // tile_dim1 = 256 rows
  g1[5] = 256;           // tensor_dim0_stride = 256 elems
  i32x4 gz = {0, 0, 0, 0};
#if __clang_major__ >= 23
  i32x8 gz8 = {0, 0, 0, 0, 0, 0, 0, 0};
  __builtin_amdgcn_tensor_load_to_lds(g0, g1, gz, gz, gz8, 0);
#else
  __builtin_amdgcn_tensor_load_to_lds(g0, g1, gz, gz, 0);
#endif
}
#endif  // HAVE_TDM

__device__ __forceinline__ v8bf cvt8(float4 a, float4 b) {
  v8bf r;
  r[0] = (__bf16)a.x; r[1] = (__bf16)a.y; r[2] = (__bf16)a.z; r[3] = (__bf16)a.w;
  r[4] = (__bf16)b.x; r[5] = (__bf16)b.y; r[6] = (__bf16)b.z; r[7] = (__bf16)b.w;
  return r;
}

// Build a 16-element fragment from two aligned 16B LDS chunks.
__device__ __forceinline__ v16bf ld_frag(const bf16_t* p0, const bf16_t* p1) {
  v8bf lo = *(const v8bf*)p0;
  v8bf hi = *(const v8bf*)p1;
  return __builtin_shufflevector(lo, hi, 0, 1, 2, 3, 4, 5, 6, 7,
                                 8, 9, 10, 11, 12, 13, 14, 15);
}

// ---------------------------------------------------------------------------
// Kernel 1a: codebook -> bf16 copy + e2[k] = sum(codebook[k]^2). Wave per row.
// ---------------------------------------------------------------------------
__global__ __launch_bounds__(256, 1)
void vq_prep_codebook(const float* __restrict__ cb, bf16_t* __restrict__ cbf,
                      float* __restrict__ e2) {
  const int lane = threadIdx.x & 31;
  const int wv   = threadIdx.x >> 5;
  const int k    = blockIdx.x * 8 + wv;
  const int d0   = lane * 8;
  const float* p = cb + (size_t)k * DDIM + d0;
  float4 a = *(const float4*)p;
  float4 b = *(const float4*)(p + 4);
  *(v8bf*)(cbf + (size_t)k * DDIM + d0) = cvt8(a, b);
  float s = a.x * a.x + a.y * a.y + a.z * a.z + a.w * a.w +
            b.x * b.x + b.y * b.y + b.z * b.z + b.w * b.w;
#pragma unroll
  for (int m = 1; m < 32; m <<= 1) s += __shfl_xor(s, m, 32);
  if (lane == 0) e2[k] = s;
}

// ---------------------------------------------------------------------------
// Kernel 1b: elementwise fp32 -> bf16 (for W_in / W_out), 8 elems per thread.
// ---------------------------------------------------------------------------
__global__ __launch_bounds__(256, 1)
void vq_cvt_bf16(const float* __restrict__ src, bf16_t* __restrict__ dst,
                 int n8) {
  const int i = blockIdx.x * blockDim.x + threadIdx.x;
  if (i < n8) {
    const float* p = src + (size_t)i * 8;
    float4 a = *(const float4*)p;
    float4 b = *(const float4*)(p + 4);
    *(v8bf*)(dst + (size_t)i * 8) = cvt8(a, b);
  }
}

// ---------------------------------------------------------------------------
// Kernel 2/6: TN GEMM  C[M,N] = A[M,KD] * B[N,KD]^T + bias, B pre-bf16.
// WG = 256 thr (8 waves, 2x4), WG tile 64x256, wave tile 32x64 (2x4 tiles).
// ---------------------------------------------------------------------------
template <int KD, bool A_F32, bool WRITE_ZBF>
__global__ __launch_bounds__(256, 1)
void vq_gemm_tn(const void* __restrict__ Ap, const bf16_t* __restrict__ Bw,
                const float* __restrict__ bias, float* __restrict__ Cf,
                bf16_t* __restrict__ Zb, int N) {
  constexpr int LDT = 40;  // padded LDS row (elems): 80B, bank-conflict-free
  __shared__ bf16_t sA[64 * LDT];
  __shared__ bf16_t sB[256 * LDT];

  const int tid  = threadIdx.x;
  const int lane = tid & 31;
  const int wv   = tid >> 5;
  const int wrow = wv & 1;    // 0..1 -> 32-row strip
  const int wcol = wv >> 1;   // 0..3 -> 64-col strip
  const int lh   = lane >> 4; // lane half
  const int lc   = lane & 15;
  const int m0 = blockIdx.x * 64;
  const int n0 = blockIdx.y * 256;

  v8f acc[2][4] = {};

  const int arow = tid >> 2;       // 64 rows, 4 threads/row
  const int ac8  = (tid & 3) * 8;  // 8-elem column chunk

  for (int k0 = 0; k0 < KD; k0 += 32) {
    // ---- stage A tile 64x32 ----
    if constexpr (A_F32) {
      const float* A = (const float*)Ap + (size_t)(m0 + arow) * KD + k0 + ac8;
      float4 f0 = *(const float4*)A;
      float4 f1 = *(const float4*)(A + 4);
      *(v8bf*)(sA + arow * LDT + ac8) = cvt8(f0, f1);
      if (k0 + 32 < KD) __builtin_prefetch(A + 32, 0, 0);
    } else {
      const bf16_t* A = (const bf16_t*)Ap + (size_t)(m0 + arow) * KD + k0 + ac8;
      cp16_lds(sA + arow * LDT + ac8, A);
    }
    // ---- stage B tile 256x32 (bf16 copy, async when available) ----
    const bf16_t* Bp = Bw + (size_t)(n0 + tid) * KD + k0;
#pragma unroll
    for (int i = 0; i < 4; i++) cp16_lds(sB + tid * LDT + i * 8, Bp + i * 8);
    cp_fence();
    __syncthreads();

    // ---- fragments (ISA 7.12.2 layouts) ----
    v16bf aF[2], bF[4];
#pragma unroll
    for (int tm = 0; tm < 2; tm++) {
      const bf16_t* p = sA + (wrow * 32 + tm * 16 + lc) * LDT + lh * 8;
      aF[tm] = ld_frag(p, p + 16);  // K chunks [lh*8, +8) and [lh*8+16, +8)
    }
#pragma unroll
    for (int tn = 0; tn < 4; tn++) {
      const bf16_t* p = sB + (wcol * 64 + tn * 16 + lc) * LDT + lh * 8;
      bF[tn] = ld_frag(p, p + 16);  // K chunks [lh*8, +8) and [lh*8+16, +8) at column lc
    }
#pragma unroll
    for (int tm = 0; tm < 2; tm++)
#pragma unroll
      for (int tn = 0; tn < 4; tn++)
        acc[tm][tn] = WMMA_BF16(aF[tm], bF[tn], acc[tm][tn]);
    __syncthreads();
  }

  // ---- epilogue: +bias, store f32 (and optional bf16 mirror of z) ----
#pragma unroll
  for (int tn = 0; tn < 4; tn++) {
    const int col = n0 + wcol * 64 + tn * 16 + lc;
    const float bv = bias[col];
#pragma unroll
    for (int tm = 0; tm < 2; tm++) {
#pragma unroll
      for (int j = 0; j < 8; j++) {
        const int m = m0 + wrow * 32 + tm * 16 + j + lh * 8;
        const float v = acc[tm][tn][j] + bv;
        Cf[(size_t)m * N + col] = v;
        if constexpr (WRITE_ZBF) Zb[(size_t)m * DDIM + col] = (bf16_t)v;
      }
    }
  }
}

// ---------------------------------------------------------------------------
// Kernel 3: fused distance GEMM + argmin.  Block owns 64 tokens, sweeps all
// 8192 codes in 256-wide chunks.  dist = e2[k] - 2*dot (z2 is constant/row).
// Codebook tiles arrive via TDM (TENSORcnt) when available.
// ---------------------------------------------------------------------------
__global__ __launch_bounds__(256, 1)
void vq_dist_argmin(const bf16_t* __restrict__ Z, const bf16_t* __restrict__ Cb,
                    const float* __restrict__ e2, int* __restrict__ idxOut) {
  constexpr int SZ_LD = 264;  // 528B row: bank stride 4 -> conflict-free
  constexpr int LDT   = 40;
  __shared__ bf16_t sZ[64 * SZ_LD];
  __shared__ bf16_t sB[256 * LDT];
  __shared__ float  sMinD[4 * 64];
  __shared__ int    sMinI[4 * 64];

  const int tid  = threadIdx.x;
  const int lane = tid & 31;
  const int wv   = tid >> 5;
  const int wrow = wv & 1;
  const int wcol = wv >> 1;
  const int lh   = lane >> 4;
  const int lc   = lane & 15;
  const int m0 = blockIdx.x * 64;

  // ---- stage the 64x256 bf16 z-tile once (async when available) ----
  for (int i = tid; i < 64 * 32; i += 256) {
    const int r = i >> 5, c = (i & 31) * 8;
    cp16_lds(sZ + r * SZ_LD + c, Z + (size_t)(m0 + r) * DDIM + c);
  }
  cp_fence();
  __syncthreads();

#if HAVE_TDM
  const unsigned sB_off = (unsigned)(unsigned long long)(const void*)sB;
#endif

  float best[2][8];
  int   bidx[2][8];
#pragma unroll
  for (int tm = 0; tm < 2; tm++)
#pragma unroll
    for (int j = 0; j < 8; j++) { best[tm][j] = 3.4e38f; bidx[tm][j] = 0; }

  for (int chunk = 0; chunk < KCB / 256; chunk++) {
    const int cbase = chunk * 256;
    v8f acc[2][4] = {};
#pragma unroll 1
    for (int ks = 0; ks < 8; ks++) {
      // ---- stage B tile 256x32 for this k-step ----
#if HAVE_TDM
      if (wv == 0) {
        tdm_load_btile(sB_off, Cb + (size_t)cbase * DDIM + ks * 32);
        __builtin_amdgcn_s_wait_tensorcnt(0);
      }
#else
      {
        const bf16_t* src = Cb + (size_t)(cbase + tid) * DDIM + ks * 32;
#pragma unroll
        for (int i = 0; i < 4; i++)
          cp16_lds(sB + tid * LDT + i * 8, src + i * 8);
        cp_fence();
      }
#endif
      __syncthreads();

      v16bf aF[2], bF[4];
#pragma unroll
      for (int tm = 0; tm < 2; tm++) {
        const bf16_t* p =
            sZ + (wrow * 32 + tm * 16 + lc) * SZ_LD + ks * 32 + lh * 8;
        aF[tm] = ld_frag(p, p + 16);
      }
#pragma unroll
      for (int tn = 0; tn < 4; tn++) {
        const bf16_t* p = sB + (wcol * 64 + tn * 16 + lc) * LDT + lh * 8;
        bF[tn] = ld_frag(p, p + 16);
      }
#pragma unroll
      for (int tm = 0; tm < 2; tm++)
#pragma unroll
        for (int tn = 0; tn < 4; tn++)
          acc[tm][tn] = WMMA_BF16(aF[tm], bF[tn], acc[tm][tn]);
      __syncthreads();
    }
    // ---- fused epilogue: running argmin (codes visited ascending/lane) ----
#pragma unroll
    for (int tn = 0; tn < 4; tn++) {
      const int code = cbase + wcol * 64 + tn * 16 + lc;
      const float ev = e2[code];
#pragma unroll
      for (int tm = 0; tm < 2; tm++)
#pragma unroll
        for (int j = 0; j < 8; j++) {
          const float d = fmaf(-2.f, acc[tm][tn][j], ev);
          if (d < best[tm][j]) { best[tm][j] = d; bidx[tm][j] = code; }
        }
    }
  }

  // ---- reduce across 16 column-lanes (per half-wave), then across waves ----
#pragma unroll
  for (int tm = 0; tm < 2; tm++)
#pragma unroll
    for (int j = 0; j < 8; j++) {
      float d = best[tm][j];
      int   ix = bidx[tm][j];
#pragma unroll
      for (int m = 1; m < 16; m <<= 1) {
        const float od = __shfl_xor(d, m, 32);
        const int   oi = __shfl_xor(ix, m, 32);
        if (od < d || (od == d && oi < ix)) { d = od; ix = oi; }
      }
      if (lc == 0) {
        const int r = wrow * 32 + tm * 16 + j + lh * 8;
        sMinD[wcol * 64 + r] = d;
        sMinI[wcol * 64 + r] = ix;
      }
    }
  __syncthreads();
  if (tid < 64) {
    float d = sMinD[tid];
    int   ix = sMinI[tid];
#pragma unroll
    for (int c = 1; c < 4; c++) {
      const float od = sMinD[c * 64 + tid];
      const int   oi = sMinI[c * 64 + tid];
      if (od < d || (od == d && oi < ix)) { d = od; ix = oi; }
    }
    idxOut[m0 + tid] = ix;
  }
}

// ---------------------------------------------------------------------------
// Kernel 4: gather codebook rows (fp32) -> q_bf16, per-token loss in fp32.
// ---------------------------------------------------------------------------
__global__ __launch_bounds__(256, 1)
void vq_gather_loss(const int* __restrict__ idx, const float* __restrict__ cb,
                    const float* __restrict__ zf, bf16_t* __restrict__ qbf,
                    float* __restrict__ tloss) {
  const int lane = threadIdx.x & 31;
  const int wv   = threadIdx.x >> 5;
  const int m    = blockIdx.x * 8 + wv;
  const int id   = idx[m];
  const int d0   = lane * 8;
  const float* c = cb + (size_t)id * DDIM + d0;
  const float* z = zf + (size_t)m * DDIM + d0;
  float4 c0 = *(const float4*)c, c1 = *(const float4*)(c + 4);
  float4 z0 = *(const float4*)z, z1 = *(const float4*)(z + 4);
  *(v8bf*)(qbf + (size_t)m * DDIM + d0) = cvt8(c0, c1);
  float s = 0.f;
  s = fmaf(z0.x - c0.x, z0.x - c0.x, s); s = fmaf(z0.y - c0.y, z0.y - c0.y, s);
  s = fmaf(z0.z - c0.z, z0.z - c0.z, s); s = fmaf(z0.w - c0.w, z0.w - c0.w, s);
  s = fmaf(z1.x - c1.x, z1.x - c1.x, s); s = fmaf(z1.y - c1.y, z1.y - c1.y, s);
  s = fmaf(z1.z - c1.z, z1.z - c1.z, s); s = fmaf(z1.w - c1.w, z1.w - c1.w, s);
#pragma unroll
  for (int msk = 1; msk < 32; msk <<= 1) s += __shfl_xor(s, msk, 32);
  if (lane == 0) tloss[m] = s;
}

// ---------------------------------------------------------------------------
// Kernel 5: deterministic fixed-order reduction of 16384 per-token losses.
// vq_loss = commit + embed = 2 * mean((z-q)^2).
// ---------------------------------------------------------------------------
__global__ __launch_bounds__(256, 1)
void vq_reduce_loss(const float* __restrict__ tloss, float* __restrict__ out) {
  __shared__ float sm[256];
  const int tid = threadIdx.x;
  float s = 0.f;
  for (int i = tid; i < BT; i += 256) s += tloss[i];
  sm[tid] = s;
  __syncthreads();
  for (int st = 128; st > 0; st >>= 1) {
    if (tid < st) sm[tid] += sm[tid + st];
    __syncthreads();
  }
  if (tid == 0) out[0] = 2.f * sm[0] / (float)((size_t)BT * DDIM);
}

// ---------------------------------------------------------------------------
extern "C" void kernel_launch(void* const* d_in, const int* in_sizes, int n_in,
                              void* d_out, int out_size, void* d_ws,
                              size_t ws_size, hipStream_t stream) {
  const float* x        = (const float*)d_in[0];  // [B,T,F]
  const float* W_in     = (const float*)d_in[1];  // [D,F]
  const float* b_in     = (const float*)d_in[2];  // [D]
  const float* codebook = (const float*)d_in[3];  // [K,D]
  const float* W_out    = (const float*)d_in[4];  // [F,D]
  const float* b_out    = (const float*)d_in[5];  // [F]

  float* out      = (float*)d_out;                              // [M,F]
  int*   idxOut   = (int*)((float*)d_out + (size_t)BT * FDIM);  // [M]
  float* lossOut  = (float*)d_out + (size_t)BT * FDIM + BT;     // scalar

  // Workspace carve-up (~38 MB).
  uint8_t* w = (uint8_t*)d_ws;
  bf16_t* cbf  = (bf16_t*)w;  w += (size_t)KCB * DDIM * sizeof(bf16_t);
  bf16_t* wbfi = (bf16_t*)w;  w += (size_t)DDIM * FDIM * sizeof(bf16_t);
  bf16_t* wbfo = (bf16_t*)w;  w += (size_t)FDIM * DDIM * sizeof(bf16_t);
  float*  e2   = (float*)w;   w += (size_t)KCB * sizeof(float);
  float*  zf   = (float*)w;   w += (size_t)BT * DDIM * sizeof(float);
  bf16_t* zbf  = (bf16_t*)w;  w += (size_t)BT * DDIM * sizeof(bf16_t);
  bf16_t* qbf  = (bf16_t*)w;  w += (size_t)BT * DDIM * sizeof(bf16_t);
  float*  tls  = (float*)w;   w += (size_t)BT * sizeof(float);

  // 1) codebook -> bf16 + e2; weights -> bf16
  vq_prep_codebook<<<KCB / 8, 256, 0, stream>>>(codebook, cbf, e2);
  vq_cvt_bf16<<<(DDIM * FDIM / 8) / 256, 256, 0, stream>>>(W_in, wbfi,
                                                           DDIM * FDIM / 8);
  vq_cvt_bf16<<<(FDIM * DDIM / 8) / 256, 256, 0, stream>>>(W_out, wbfo,
                                                           FDIM * DDIM / 8);
  // 2) z = x @ W_in^T + b_in  (f32 + bf16 mirror)
  vq_gemm_tn<FDIM, true, true><<<dim3(BT / 64, 1), 256, 0, stream>>>(
      (const void*)x, wbfi, b_in, zf, zbf, DDIM);
  // 3) fused distance GEMM + argmin -> indices
  vq_dist_argmin<<<BT / 64, 256, 0, stream>>>(zbf, cbf, e2, idxOut);
  // 4) gather q (bf16) + exact fp32 per-token loss
  vq_gather_loss<<<BT / 8, 256, 0, stream>>>(idxOut, codebook, zf, qbf, tls);
  // 5) vq_loss scalar
  vq_reduce_loss<<<1, 256, 0, stream>>>(tls, lossOut);
  // 6) q_features = q @ W_out^T + b_out
  vq_gemm_tn<DDIM, false, false><<<dim3(BT / 64, FDIM / 256), 256, 0, stream>>>(
      (const void*)qbf, wbfo, b_out, out, nullptr, FDIM);
}
